// Dynamic_conv2d_12360915878240
// MI455X (gfx1250) — matrix operand order repaired and checked by reading, not compiled, not run on hardware
//
#include <hip/hip_runtime.h>

typedef __attribute__((ext_vector_type(16))) __bf16 v16bf;
typedef __attribute__((ext_vector_type(8)))  float  v8f;

union ABFrag {
    uint4 u[2];
    v16bf v;
};

__device__ __forceinline__ unsigned short f2bf(float f) {
    unsigned int u = __float_as_uint(f);
    unsigned int r = (u + 0x7FFFu + ((u >> 16) & 1u)) >> 16;   // round-to-nearest-even
    return (unsigned short)r;
}

// Async global->LDS 128b copy (CDNA5, ASYNCcnt-tracked).
__device__ __forceinline__ void async_g2l_b128(unsigned lds_off, const void* gptr) {
    asm volatile("global_load_async_to_lds_b128 %0, %1, off"
                 :: "v"(lds_off), "v"(gptr) : "memory");
}
__device__ __forceinline__ void wait_asynccnt0() {
    asm volatile("s_wait_asynccnt 0x0" ::: "memory");
}

// ---------------------------------------------------------------------------
// Prologue 1: agg_w[b][pos][co][ci] = bf16( sum_k attn[b,k] * W[k,co,ci,kh,kw] )
// ---------------------------------------------------------------------------
__global__ void agg_w_kernel(const float* __restrict__ w,
                             const float* __restrict__ attn,
                             unsigned short* __restrict__ ws_w) {
    int idx = blockIdx.x * 256 + threadIdx.x;
    if (idx >= 16 * 9 * 64 * 64) return;
    int b   = idx / 36864;
    int r   = idx % 36864;
    int pos = r / 4096;
    int r2  = r % 4096;
    int co  = r2 >> 6;
    int ci  = r2 & 63;
    float s = 0.f;
#pragma unroll
    for (int k = 0; k < 5; ++k)
        s += attn[b * 5 + k] * w[((k * 64 + co) * 64 + ci) * 9 + pos];
    ws_w[idx] = f2bf(s);
}

// Prologue 2: agg_b[b][co] = sum_k attn[b,k] * bias[k,co]   (f32)
__global__ void agg_b_kernel(const float* __restrict__ attn,
                             const float* __restrict__ bias,
                             float* __restrict__ ws_b) {
    int t = blockIdx.x * 256 + threadIdx.x;
    if (t >= 16 * 64) return;
    int b = t >> 6, co = t & 63;
    float s = 0.f;
#pragma unroll
    for (int k = 0; k < 5; ++k) s += attn[b * 5 + k] * bias[k * 64 + co];
    ws_b[t] = s;
}

// ---------------------------------------------------------------------------
// Prologue 3: x[b][ci][h][w] f32  ->  ws_x[b][h+1][w+1][ci] bf16 (194x194 padded)
// ---------------------------------------------------------------------------
__global__ __launch_bounds__(256)
void xpose_kernel(const float* __restrict__ x, unsigned short* __restrict__ ws_x) {
    __shared__ __align__(16) unsigned short s[64 * 72];
    const int t  = threadIdx.x;
    const int w0 = blockIdx.x * 64;   // 0,64,128
    const int h  = blockIdx.y;        // 0..191
    const int b  = blockIdx.z;        // 0..15
    const float* xb = x + ((size_t)b * 64 * 192 + h) * 192 + w0;
#pragma unroll
    for (int j = 0; j < 16; ++j) {
        int idx = j * 256 + t;        // 4096 = 64 ci x 64 w
        int ci = idx >> 6, w = idx & 63;
        s[w * 72 + ci] = f2bf(xb[(size_t)ci * 192 * 192 + w]);
    }
    __syncthreads();
    unsigned short* ob = ws_x + (((size_t)b * 194 + (h + 1)) * 194 + (w0 + 1)) * 64;
#pragma unroll
    for (int j = 0; j < 2; ++j) {
        int e = j * 256 + t;          // 512 uint4 chunks
        int w = e >> 3, c8 = (e & 7) * 8;
        *(uint4*)&ob[(size_t)w * 64 + c8] = *(const uint4*)&s[w * 72 + c8];
    }
}

// Zero the 1-pixel border of the padded bf16 buffer (conv pad=1).
__global__ void xpad_kernel(unsigned short* __restrict__ ws_x) {
    const int t = threadIdx.x;
    unsigned short* xb = ws_x + (size_t)blockIdx.x * 194 * 194 * 64;
#pragma unroll 1
    for (int i = 0; i < 25; ++i) {
        int e = i * 256 + t;          // 776 border px * 8 chunks = 6208
        if (e >= 6208) break;
        int p = e >> 3, c8 = (e & 7) * 8;
        int side = p / 194, q = p % 194;
        size_t pix;
        if (side == 0)      pix = (size_t)q;                  // h = 0
        else if (side == 1) pix = (size_t)193 * 194 + q;      // h = 193
        else if (side == 2) pix = (size_t)q * 194;            // w = 0
        else                pix = (size_t)q * 194 + 193;      // w = 193
        *(uint4*)&xb[pix * 64 + c8] = (uint4){0u, 0u, 0u, 0u};
    }
}

// ---------------------------------------------------------------------------
// Main kernel (async path): implicit-GEMM conv, v_wmma_f32_16x16x32_bf16.
// Wave tile = 2 M-tiles x 4 N-tiles. Per K-step: issue all 12 ds_load_b128
// first (2 A frags + 4 B frags in distinct registers), then an 8-WMMA burst
// so LDS latency hides behind the matrix pipe. Weights double-buffered via
// async DMA (global_load_async_to_lds_b128 / s_wait_asynccnt).
// ---------------------------------------------------------------------------
#define XSTR 72
#define WSTR 72

__global__ __launch_bounds__(256)
void dynconv_async_kernel(const unsigned short* __restrict__ ws_x,
                          const unsigned short* __restrict__ ws_w,
                          const float* __restrict__ ws_b,
                          float* __restrict__ out) {
    __shared__ __align__(16) unsigned short s_x[324 * XSTR];      // 18x18 halo [pix][ci]
    __shared__ __align__(16) unsigned short s_w[2][64 * WSTR];    // double-buffered [co][ci]

    const int t    = threadIdx.x;
    const int lane = t & 31;
    const int wave = t >> 5;
    const int wq   = wave & 3;    // N quarter: n-tiles wq*4 .. wq*4+3
    const int wp   = wave >> 2;   // M pair:   co-tiles wp*2, wp*2+1
    const int mrow = lane & 15;
    const int half = lane >> 4;
    const int n    = lane & 15;

    const int gx0 = blockIdx.x * 16;
    const int gy0 = blockIdx.y * 16;
    const int b   = blockIdx.z;

    const unsigned short* xb = ws_x + (size_t)b * 194 * 194 * 64;
    const unsigned short* wb = ws_w + (size_t)b * 9 * 4096;

    // ---- issue weight slice for pos=0 (async) ----
#pragma unroll
    for (int i = 0; i < 2; ++i) {
        int e = i * 256 + t;                     // 512 chunks of 8 bf16
        int co = e >> 3, ci = (e & 7) * 8;
        async_g2l_b128((unsigned)(size_t)&s_w[0][co * WSTR + ci], wb + e * 8);
    }
    // ---- issue halo (18x18x64 bf16 = 2592 chunks, async) ----
#pragma unroll 1
    for (int i = 0; i < 11; ++i) {
        int e = i * 256 + t;
        if (e < 2592) {
            int p = e >> 3, c8 = (e & 7) * 8;
            int iy = p / 18, ix = p % 18;
            const unsigned short* gp = xb + (((size_t)(gy0 + iy) * 194) + gx0 + ix) * 64 + c8;
            async_g2l_b128((unsigned)(size_t)&s_x[p * XSTR + c8], gp);
        }
    }
    wait_asynccnt0();
    __syncthreads();

    v8f acc[2][4];
#pragma unroll
    for (int i = 0; i < 2; ++i)
#pragma unroll
        for (int j = 0; j < 4; ++j)
            acc[i][j] = (v8f){0.f, 0.f, 0.f, 0.f, 0.f, 0.f, 0.f, 0.f};

#pragma unroll 1
    for (int pos = 0; pos < 9; ++pos) {
        const int kh = pos / 3, kw = pos % 3;
        const int buf = pos & 1;

        // prefetch next weight slice into the other buffer (overlaps WMMAs)
        if (pos < 8) {
            const unsigned short* wnext = wb + (size_t)(pos + 1) * 4096;
#pragma unroll
            for (int i = 0; i < 2; ++i) {
                int e = i * 256 + t;
                int co = e >> 3, ci = (e & 7) * 8;
                async_g2l_b128((unsigned)(size_t)&s_w[buf ^ 1][co * WSTR + ci], wnext + e * 8);
            }
        }

#pragma unroll
        for (int ks = 0; ks < 2; ++ks) {
            // --- issue ALL fragment loads for this K-step first ---
            ABFrag a[2];
#pragma unroll
            for (int i = 0; i < 2; ++i) {
                const unsigned short* wrow =
                    &s_w[buf][((wp * 2 + i) * 16 + mrow) * WSTR + ks * 32];
                a[i].u[0] = *(const uint4*)&wrow[half * 16];
                a[i].u[1] = *(const uint4*)&wrow[half * 16 + 8];
            }
            ABFrag bbv[4];
#pragma unroll
            for (int j = 0; j < 4; ++j) {
                const int pix = (wq * 4 + j + kh) * 18 + (n + kw);
                const unsigned short* xrow = &s_x[pix * XSTR + ks * 32 + half * 16];
                bbv[j].u[0] = *(const uint4*)&xrow[0];
                bbv[j].u[1] = *(const uint4*)&xrow[8];
            }
            // --- 8-WMMA burst ---
#pragma unroll
            for (int j = 0; j < 4; ++j) {
                acc[0][j] = __builtin_amdgcn_wmma_f32_16x16x32_bf16(
                    false, a[0].v, false, bbv[j].v, (short)0, acc[0][j], false, false);
                acc[1][j] = __builtin_amdgcn_wmma_f32_16x16x32_bf16(
                    false, a[1].v, false, bbv[j].v, (short)0, acc[1][j], false, false);
            }
        }
        wait_asynccnt0();
        __syncthreads();
    }

    // ---- epilogue: bias add + coalesced f32 stores ----
    float* ob = out + (size_t)b * 64 * 192 * 192;
#pragma unroll
    for (int i = 0; i < 2; ++i) {
        const int co_base = (wp * 2 + i) * 16 + half * 8;
        float bv[8];
#pragma unroll
        for (int v = 0; v < 8; ++v) bv[v] = ws_b[b * 64 + co_base + v];
#pragma unroll
        for (int j = 0; j < 4; ++j) {
            const int gy = gy0 + wq * 4 + j;
            const int gx = gx0 + n;
#pragma unroll
            for (int v = 0; v < 8; ++v)
                ob[((co_base + v) * 192 + gy) * 192 + gx] = acc[i][j][v] + bv[v];
        }
    }
}

// ---------------------------------------------------------------------------
// Fallback kernel (small ws): stages f32 halo directly, converts in-kernel.
// ---------------------------------------------------------------------------
__global__ __launch_bounds__(256)
void dynconv_wmma_kernel(const float* __restrict__ x,
                         const unsigned short* __restrict__ ws_w,
                         const float* __restrict__ ws_b,
                         float* __restrict__ out) {
    __shared__ __align__(16) unsigned short s_x[324 * XSTR];
    __shared__ __align__(16) unsigned short s_w[64 * WSTR];

    const int t    = threadIdx.x;
    const int lane = t & 31;
    const int wave = t >> 5;
    const int wm   = wave >> 1;
    const int wn   = wave & 1;
    const int mrow = lane & 15;
    const int half = lane >> 4;
    const int n    = lane & 15;

    const int gx0 = blockIdx.x * 16;
    const int gy0 = blockIdx.y * 16;
    const int b   = blockIdx.z;

    const float* xb = x + (size_t)b * 64 * 192 * 192;
#pragma unroll 1
    for (int i = 0; i < 81; ++i) {
        int idx = i * 256 + t;
        int ci  = idx / 324;
        int p   = idx % 324;
        int iy  = p / 18, ix = p % 18;
        int h   = gy0 + iy - 1;
        int wcl = gx0 + ix - 1;
        float v = 0.f;
        if (h >= 0 && h < 192 && wcl >= 0 && wcl < 192)
            v = xb[(ci * 192 + h) * 192 + wcl];
        s_x[p * XSTR + ci] = f2bf(v);
    }

    v8f acc[8];
#pragma unroll
    for (int nt = 0; nt < 8; ++nt) acc[nt] = (v8f){0.f, 0.f, 0.f, 0.f, 0.f, 0.f, 0.f, 0.f};

#pragma unroll 1
    for (int pos = 0; pos < 9; ++pos) {
        const int kh = pos / 3, kw = pos % 3;
        __syncthreads();
        const uint4* gsrc = (const uint4*)(ws_w + ((size_t)(b * 9 + pos)) * 4096);
#pragma unroll
        for (int i = 0; i < 2; ++i) {
            int e = i * 256 + t;
            int co = (e * 8) >> 6, ci = (e * 8) & 63;
            *(uint4*)&s_w[co * WSTR + ci] = gsrc[e];
        }
        __syncthreads();
#pragma unroll
        for (int ks = 0; ks < 2; ++ks) {
            ABFrag a;
            const unsigned short* wrow = &s_w[(wm * 16 + mrow) * WSTR + ks * 32];
            a.u[0] = *(const uint4*)&wrow[half * 16];
            a.u[1] = *(const uint4*)&wrow[half * 16 + 8];
#pragma unroll
            for (int nt = 0; nt < 8; ++nt) {
                const int pix = (wn * 8 + nt + kh) * 18 + (n + kw);
                ABFrag bb;
                const unsigned short* xrow = &s_x[pix * XSTR + ks * 32 + half * 16];
                bb.u[0] = *(const uint4*)&xrow[0];
                bb.u[1] = *(const uint4*)&xrow[8];
                acc[nt] = __builtin_amdgcn_wmma_f32_16x16x32_bf16(
                    false, a.v, false, bb.v, (short)0, acc[nt], false, false);
            }
        }
    }

    const int co_base = wm * 16 + half * 8;
    float bv[8];
#pragma unroll
    for (int v = 0; v < 8; ++v) bv[v] = ws_b[b * 64 + co_base + v];
    float* ob = out + (size_t)b * 64 * 192 * 192;
#pragma unroll
    for (int nt = 0; nt < 8; ++nt) {
        const int gy = gy0 + wn * 8 + nt;
        const int gx = gx0 + n;
#pragma unroll
        for (int v = 0; v < 8; ++v)
            ob[((co_base + v) * 192 + gy) * 192 + gx] = acc[nt][v] + bv[v];
    }
}

// ---------------------------------------------------------------------------
extern "C" void kernel_launch(void* const* d_in, const int* in_sizes, int n_in,
                              void* d_out, int out_size, void* d_ws, size_t ws_size,
                              hipStream_t stream) {
    const float* x    = (const float*)d_in[0];   // [16,64,192,192]
    const float* attn = (const float*)d_in[1];   // [16,5]
    const float* w    = (const float*)d_in[2];   // [5,64,64,3,3]
    const float* bias = (const float*)d_in[3];   // [5,64]
    float* out        = (float*)d_out;           // [16,64,192,192]

    const size_t WS_W_BYTES = (size_t)16 * 9 * 64 * 64 * 2;   // 1,179,648
    const size_t WS_B_OFF   = WS_W_BYTES;
    const size_t WS_X_OFF   = WS_B_OFF + 4096;                // 16B aligned
    const size_t WS_X_BYTES = (size_t)16 * 194 * 194 * 64 * 2;

    unsigned short* ws_w = (unsigned short*)d_ws;
    float* ws_b          = (float*)((char*)d_ws + WS_B_OFF);
    unsigned short* ws_x = (unsigned short*)((char*)d_ws + WS_X_OFF);

    agg_w_kernel<<<2304, 256, 0, stream>>>(w, attn, ws_w);
    agg_b_kernel<<<4, 256, 0, stream>>>(attn, bias, ws_b);

    dim3 grid(12, 12, 16);
    if (ws_size >= WS_X_OFF + WS_X_BYTES) {
        xpose_kernel<<<dim3(3, 192, 16), 256, 0, stream>>>(x, ws_x);
        xpad_kernel<<<16, 256, 0, stream>>>(ws_x);
        dynconv_async_kernel<<<grid, 256, 0, stream>>>(ws_x, ws_w, ws_b, out);
    } else {
        dynconv_wmma_kernel<<<grid, 256, 0, stream>>>(x, ws_w, ws_b, out);
    }
}
